// VAE_54692113547524
// MI455X (gfx1250) — compile-verified, matrix operand order checked by reading, not run on hardware
//
#include <hip/hip_runtime.h>
#include <hip/hip_bf16.h>
#include <math.h>

// Problem constants (match reference)
#define B_  2048
#define T_  29
#define H_  1024
#define Z_  1024
#define V1_ 51   // V+1

typedef __bf16 bf16;
typedef __attribute__((ext_vector_type(16))) __bf16 v16bf;
typedef __attribute__((ext_vector_type(8)))  float  v8f;

__device__ __forceinline__ float sigmoidf_(float x) { return 1.0f / (1.0f + __expf(-x)); }

// ---------------------------------------------------------------------------
// CDNA5 async global->LDS copy (16B per lane), ASYNCcnt-tracked.
// Inline asm per ISA 15.18.3: VDST = LDS byte address, VADDR = 64-bit global
// address (GV mode, SADDR = off).
// ---------------------------------------------------------------------------
__device__ __forceinline__ void async_cp16(const bf16* g, bf16* l) {
    asm volatile("global_load_async_to_lds_b128 %0, %1, off"
                 :: "v"((uint32_t)(uintptr_t)l), "v"((uint64_t)(uintptr_t)g)
                 : "memory");
}

__device__ __forceinline__ void wait_async0() {
    asm volatile("s_wait_asynccnt 0x0" ::: "memory");
}

// ---------------------------------------------------------------------------
// Generic NT GEMM:  C[M,N] = act( A[M,K] * W[N,K]^T + Cinit + bias )
// A, W bf16 row-major; f32 accumulate via v_wmma_f32_16x16x32_bf16.
// Block: 256 threads (8 waves). Block tile 128(M) x 128(N), K-step 32,
// K-loop unrolled x2 so the LDS double-buffer index is compile-time.
// Wave (wm,wn) in 2x4 grid owns 64(M) x 32(N): accum[4][2] 16x16 f32 tiles.
// Requires: M % 128 == 0, K % 64 == 0, and W readable for full 128-row tiles
// (pad W with zero rows when N % 128 != 0). Stores guarded by true M,N.
// ---------------------------------------------------------------------------
#define LDS_STRIDE 40   // bf16 elements per row (80B): bank-conflict-free b128

__global__ __launch_bounds__(256) void gemm_bf16_wmma(
    const bf16* __restrict__ A, int lda,
    const bf16* __restrict__ W, int ldb,
    const float* __restrict__ Cinit, int ldci,
    const float* __restrict__ bias,
    float* __restrict__ outF, int ldof,
    bf16* __restrict__ outB, int ldob,
    int M, int N, int K, int relu)
{
    __shared__ bf16 Abuf[2][128 * LDS_STRIDE];
    __shared__ bf16 Bbuf[2][128 * LDS_STRIDE];

    const int tid    = threadIdx.x;
    const int wave   = tid >> 5;
    const int lane   = tid & 31;
    const int hi     = lane >> 4;      // half-wave select
    const int lanelo = lane & 15;

    const int wm = wave & 1;
    const int wn = wave >> 1;
    const int m0 = blockIdx.x * 128 + wm * 64;  // wave M base (4 subtiles of 16)
    const int n0 = blockIdx.y * 128 + wn * 32;  // wave N base (2 subtiles of 16)

    const bf16* Ablk = A + (size_t)(blockIdx.x * 128) * lda;
    const bf16* Bblk = W + (size_t)(blockIdx.y * 128) * ldb;

    v8f acc[4][2];
    #pragma unroll
    for (int i = 0; i < 4; ++i)
        #pragma unroll
        for (int j = 0; j < 2; ++j)
            #pragma unroll
            for (int v = 0; v < 8; ++v) {
                const int m = m0 + i * 16 + v + 8 * hi;
                const int n = n0 + j * 16 + lanelo;
                float cv = 0.0f;
                if (Cinit && m < M && n < N) cv = Cinit[(size_t)m * ldci + n];
                acc[i][j][v] = cv;
            }

    // stage one 128x32 bf16 tile into LDS: 512 chunks of 16B, 2 per thread,
    // both unconditionally valid (tid<256) -> no predication.
    const int srow = tid >> 2;
    const int sko  = (tid & 3) * 8;   // bf16 elements
    auto stage = [&](const bf16* src, int ld, bf16* dst) {
        async_cp16(src + (size_t)srow * ld + sko,        dst + srow * LDS_STRIDE + sko);
        async_cp16(src + (size_t)(srow + 64) * ld + sko, dst + (srow + 64) * LDS_STRIDE + sko);
    };

    // 16-bit fragment layout (ISA 7.12.2): lane l holds row/col (l&15);
    // element e -> k = (e&7) + 16*(e>>3) + 8*(l>>4)
    // => two contiguous 16B LDS reads per fragment per lane.
    auto compute = [&](const bf16* Ab, const bf16* Bb) {
        v16bf afrag[4];
        #pragma unroll
        for (int i = 0; i < 4; ++i) {
            const bf16* ap = &Ab[(wm * 64 + i * 16 + lanelo) * LDS_STRIDE + hi * 8];
            union { uint4 u[2]; v16bf v; } un;
            un.u[0] = *reinterpret_cast<const uint4*>(ap);
            un.u[1] = *reinterpret_cast<const uint4*>(ap + 16);
            afrag[i] = un.v;
        }
        v16bf bfrag[2];
        #pragma unroll
        for (int j = 0; j < 2; ++j) {
            const bf16* bp = &Bb[(wn * 32 + j * 16 + lanelo) * LDS_STRIDE + hi * 8];
            union { uint4 u[2]; v16bf v; } un;
            un.u[0] = *reinterpret_cast<const uint4*>(bp);
            un.u[1] = *reinterpret_cast<const uint4*>(bp + 16);
            bfrag[j] = un.v;
        }
        #pragma unroll
        for (int i = 0; i < 4; ++i)
            #pragma unroll
            for (int j = 0; j < 2; ++j)
                acc[i][j] = __builtin_amdgcn_wmma_f32_16x16x32_bf16(
                    false, afrag[i], false, bfrag[j], (short)0, acc[i][j], false, false);
    };

    stage(Ablk, lda, Abuf[0]);
    stage(Bblk, ldb, Bbuf[0]);
    wait_async0();
    __syncthreads();

    // K % 64 == 0 at every call site; double-buffer index is compile-time.
    for (int k0 = 0; k0 < K; k0 += 64) {
        // k0+32 < K always (K multiple of 64): unconditional prefetch
        stage(Ablk + k0 + 32, lda, Abuf[1]);
        stage(Bblk + k0 + 32, ldb, Bbuf[1]);
        compute(Abuf[0], Bbuf[0]);
        wait_async0();
        __syncthreads();

        if (k0 + 64 < K) {
            stage(Ablk + k0 + 64, lda, Abuf[0]);
            stage(Bblk + k0 + 64, ldb, Bbuf[0]);
        }
        compute(Abuf[1], Bbuf[1]);
        wait_async0();
        __syncthreads();
    }

    #pragma unroll
    for (int i = 0; i < 4; ++i)
        #pragma unroll
        for (int j = 0; j < 2; ++j) {
            const int n = n0 + j * 16 + lanelo;
            const float bv = (bias && n < N) ? bias[n] : 0.0f;
            #pragma unroll
            for (int v = 0; v < 8; ++v) {
                const int m = m0 + i * 16 + v + 8 * hi;
                if (m < M && n < N) {
                    float r = acc[i][j][v] + bv;
                    if (relu) r = fmaxf(r, 0.0f);
                    if (outF) outF[(size_t)m * ldof + n] = r;
                    if (outB) outB[(size_t)m * ldob + n] = (bf16)r;
                }
            }
        }
}

// ---------------------------------------------------------------------------
// Elementwise / support kernels
// ---------------------------------------------------------------------------
__global__ void f32_to_bf16_k(const float* __restrict__ in, bf16* __restrict__ out, long n) {
    long i = (long)blockIdx.x * 256 + threadIdx.x;
    if (i < n) out[i] = (bf16)in[i];
}

__global__ void add_vec_k(const float* __restrict__ a, const float* __restrict__ b,
                          float* __restrict__ out, int n) {
    int i = blockIdx.x * 256 + threadIdx.x;
    if (i < n) out[i] = a[i] + b[i];
}

__global__ void fill_bf16_zero_k(bf16* __restrict__ p, long n) {
    long i = (long)blockIdx.x * 256 + threadIdx.x;
    if (i < n) p[i] = (bf16)0.0f;
}

__global__ void fill_f32_zero_k(float* __restrict__ p, long n) {
    long i = (long)blockIdx.x * 256 + threadIdx.x;
    if (i < n) p[i] = 0.0f;
}

// pooled[b,h] = mean_t emb[x[b,t], h]  -> bf16
__global__ void embed_pool_k(const int* __restrict__ x, const float* __restrict__ emb,
                             bf16* __restrict__ pooled) {
    long idx = (long)blockIdx.x * 256 + threadIdx.x;
    if (idx >= (long)B_ * H_) return;
    const int b = (int)(idx / H_);
    const int h = (int)(idx % H_);
    float s = 0.0f;
    #pragma unroll 1
    for (int t = 0; t < T_; ++t) {
        const int tok = x[b * T_ + t];
        s += emb[(size_t)tok * H_ + h];
    }
    pooled[idx] = (bf16)(s * (1.0f / (float)T_));
}

// z = mu + eps * exp(0.5*log_var); emits mu/log_var to d_out, z in bf16
__global__ void reparam_k(const float* __restrict__ henc, const float* __restrict__ eps,
                          bf16* __restrict__ zbf, float* __restrict__ mu_out,
                          float* __restrict__ lv_out) {
    long i = (long)blockIdx.x * 256 + threadIdx.x;
    if (i >= (long)B_ * Z_) return;
    const int b = (int)(i / Z_);
    const int n = (int)(i % Z_);
    const float mu = henc[(size_t)b * (2 * Z_) + n];
    const float lv = henc[(size_t)b * (2 * Z_) + Z_ + n];
    const float z  = mu + eps[i] * __expf(0.5f * lv);
    zbf[i]    = (bf16)z;
    mu_out[i] = mu;
    lv_out[i] = lv;
}

// torch gate order i,f,g,o; gates already include x_proj (folded via Cinit)
__global__ void lstm_cell_k(const float* __restrict__ gates, float* __restrict__ c,
                            bf16* __restrict__ hbf) {
    long idx = (long)blockIdx.x * 256 + threadIdx.x;
    if (idx >= (long)B_ * H_) return;
    const int b = (int)(idx / H_);
    const int j = (int)(idx % H_);
    const float* g = gates + (size_t)b * (4 * H_);
    const float ig = sigmoidf_(g[j]);
    const float fg = sigmoidf_(g[j + H_]);
    const float gg = tanhf(g[j + 2 * H_]);
    const float og = sigmoidf_(g[j + 3 * H_]);
    const float cn = fg * c[idx] + ig * gg;
    c[idx]   = cn;
    hbf[idx] = (bf16)(og * tanhf(cn));
}

// in-place log_softmax over rows of width V1_ (one wave32 per row)
__global__ __launch_bounds__(256) void log_softmax_k(float* __restrict__ logits, long rows) {
    const long wrow = ((long)blockIdx.x * 256 + threadIdx.x) >> 5;
    const int  lane = threadIdx.x & 31;
    if (wrow >= rows) return;
    float* row = logits + wrow * V1_;
    const float v0 = (lane < V1_)       ? row[lane]       : -INFINITY;
    const float v1 = (lane + 32 < V1_)  ? row[lane + 32]  : -INFINITY;
    float m = fmaxf(v0, v1);
    #pragma unroll
    for (int off = 16; off > 0; off >>= 1) m = fmaxf(m, __shfl_xor(m, off, 32));
    float s = ((lane < V1_) ? __expf(v0 - m) : 0.0f) +
              ((lane + 32 < V1_) ? __expf(v1 - m) : 0.0f);
    #pragma unroll
    for (int off = 16; off > 0; off >>= 1) s += __shfl_xor(s, off, 32);
    const float lse = m + __logf(s);
    if (lane < V1_)      row[lane]      = v0 - lse;
    if (lane + 32 < V1_) row[lane + 32] = v1 - lse;
}

// ---------------------------------------------------------------------------
// Host orchestration
// ---------------------------------------------------------------------------
extern "C" void kernel_launch(void* const* d_in, const int* in_sizes, int n_in,
                              void* d_out, int out_size, void* d_ws, size_t ws_size,
                              hipStream_t stream) {
    (void)in_sizes; (void)n_in; (void)out_size; (void)ws_size;

    const int*   x      = (const int*)  d_in[0];
    const float* eps    = (const float*)d_in[1];
    const float* emb    = (const float*)d_in[2];
    const float* enc_w1 = (const float*)d_in[3];
    const float* enc_b1 = (const float*)d_in[4];
    const float* enc_w2 = (const float*)d_in[5];
    const float* enc_b2 = (const float*)d_in[6];
    const float* w_ih   = (const float*)d_in[7];
    const float* w_hh   = (const float*)d_in[8];
    const float* b_ih   = (const float*)d_in[9];
    const float* b_hh   = (const float*)d_in[10];
    const float* out_w  = (const float*)d_in[11];
    const float* out_b  = (const float*)d_in[12];

    float* logits = (float*)d_out;                          // [B,T,V1]
    float* mu_out = logits + (size_t)B_ * T_ * V1_;         // [B,Z]
    float* lv_out = mu_out + (size_t)B_ * Z_;               // [B,Z]

    // workspace carve (aligned 256B)
    char* wsb = (char*)d_ws;
    size_t off = 0;
    auto alloc = [&](size_t bytes) -> void* {
        void* p = wsb + off;
        off = (off + bytes + 255) & ~(size_t)255;
        return p;
    };
    bf16*  w1b    = (bf16*) alloc((size_t)H_ * H_ * 2);        // enc_w1
    bf16*  w2b    = (bf16*) alloc((size_t)2 * Z_ * H_ * 2);    // enc_w2
    bf16*  wihb   = (bf16*) alloc((size_t)4 * H_ * H_ * 2);    // w_ih
    bf16*  whhb   = (bf16*) alloc((size_t)4 * H_ * H_ * 2);    // w_hh
    bf16*  owb    = (bf16*) alloc((size_t)128 * H_ * 2);       // out_w padded to 128 rows
    float* bsum   = (float*)alloc((size_t)4 * H_ * 4);         // b_ih + b_hh
    bf16*  pooled = (bf16*) alloc((size_t)B_ * H_ * 2);
    bf16*  h1b    = (bf16*) alloc((size_t)B_ * H_ * 2);
    float* henc   = (float*)alloc((size_t)B_ * 2 * Z_ * 4);
    bf16*  zb     = (bf16*) alloc((size_t)B_ * Z_ * 2);
    float* xproj  = (float*)alloc((size_t)B_ * 4 * H_ * 4);
    bf16*  hb     = (bf16*) alloc((size_t)B_ * H_ * 2);
    float* cstate = (float*)alloc((size_t)B_ * H_ * 4);
    float* gates  = (float*)alloc((size_t)B_ * 4 * H_ * 4);

    auto cvt = [&](const float* src, bf16* dst, long n) {
        f32_to_bf16_k<<<dim3((unsigned)((n + 255) / 256)), dim3(256), 0, stream>>>(src, dst, n);
    };
    auto gemm = [&](const bf16* A, int lda, const bf16* W, int ldb,
                    const float* Ci, int ldci, const float* bias,
                    float* oF, int ldof, bf16* oB, int ldob,
                    int M, int N, int K, int relu) {
        dim3 g((unsigned)(M / 128), (unsigned)((N + 127) / 128));
        gemm_bf16_wmma<<<g, dim3(256), 0, stream>>>(A, lda, W, ldb, Ci, ldci, bias,
                                                    oF, ldof, oB, ldob, M, N, K, relu);
    };

    // 1) weight conversion (bf16, L2-resident thereafter) + bias fold.
    //    out_w padded with zero rows to 128 so GEMM tiles never go ragged.
    cvt(enc_w1, w1b,  (long)H_ * H_);
    cvt(enc_w2, w2b,  (long)2 * Z_ * H_);
    cvt(w_ih,   wihb, (long)4 * H_ * H_);
    cvt(w_hh,   whhb, (long)4 * H_ * H_);
    {
        long n = (long)128 * H_;
        fill_bf16_zero_k<<<dim3((unsigned)((n + 255) / 256)), dim3(256), 0, stream>>>(owb, n);
    }
    cvt(out_w, owb, (long)V1_ * H_);
    add_vec_k<<<dim3((4 * H_ + 255) / 256), dim3(256), 0, stream>>>(b_ih, b_hh, bsum, 4 * H_);

    // 2) embed + mean pool
    {
        long n = (long)B_ * H_;
        embed_pool_k<<<dim3((unsigned)((n + 255) / 256)), dim3(256), 0, stream>>>(x, emb, pooled);
    }

    // 3) encoder: h1 = relu(pooled @ enc_w1^T + b1)
    gemm(pooled, H_, w1b, H_, nullptr, 0, enc_b1, nullptr, 0, h1b, H_, B_, H_, H_, 1);
    //    henc = h1 @ enc_w2^T + b2
    gemm(h1b, H_, w2b, H_, nullptr, 0, enc_b2, henc, 2 * Z_, nullptr, 0, B_, 2 * Z_, H_, 0);

    // 4) reparameterize; mu/log_var -> d_out
    {
        long n = (long)B_ * Z_;
        reparam_k<<<dim3((unsigned)((n + 255) / 256)), dim3(256), 0, stream>>>(henc, eps, zb, mu_out, lv_out);
    }

    // 5) x_proj = z @ w_ih^T + (b_ih + b_hh)   (computed once)
    gemm(zb, Z_, wihb, Z_, nullptr, 0, bsum, xproj, 4 * H_, nullptr, 0, B_, 4 * H_, Z_, 0);

    // 6) h0 = c0 = 0
    {
        long n = (long)B_ * H_;
        fill_bf16_zero_k<<<dim3((unsigned)((n + 255) / 256)), dim3(256), 0, stream>>>(hb, n);
        fill_f32_zero_k <<<dim3((unsigned)((n + 255) / 256)), dim3(256), 0, stream>>>(cstate, n);
    }

    // 7) LSTM recurrence + fused per-step output projection
    for (int t = 0; t < T_; ++t) {
        // gates = x_proj + h @ w_hh^T   (x_proj folded in as accumulator init)
        gemm(hb, H_, whhb, H_, xproj, 4 * H_, nullptr, gates, 4 * H_, nullptr, 0,
             B_, 4 * H_, H_, 0);
        // pointwise cell: c,h update
        {
            long n = (long)B_ * H_;
            lstm_cell_k<<<dim3((unsigned)((n + 255) / 256)), dim3(256), 0, stream>>>(gates, cstate, hb);
        }
        // logits[:, t, :] = h @ out_w^T + out_b   (row stride T*V1; W padded to 128 rows)
        gemm(hb, H_, owb, H_, nullptr, 0, out_b, logits + (size_t)t * V1_, T_ * V1_,
             nullptr, 0, B_, V1_, H_, 0);
    }

    // 8) in-place log_softmax over all B*T rows
    {
        long rows = (long)B_ * T_;
        long waves_per_block = 8;
        long blocks = (rows + waves_per_block - 1) / waves_per_block;
        log_softmax_k<<<dim3((unsigned)blocks), dim3(256), 0, stream>>>(logits, rows);
    }
}
